// serverGIN_20212116095377
// MI455X (gfx1250) — matrix operand order repaired and checked by reading, not compiled, not run on hardware
//
#include <hip/hip_runtime.h>

// ---------------------------------------------------------------------------
// GIN forward for MI455X (gfx1250, wave32).
//  - edge gather/scatter: f32, float4 loads + global_atomic_add_f32 (memory-bound phase)
//  - MLP GEMMs: LDS-staged bf16 tiles -> v_wmma_f32_16x16x32_bf16, bias+ReLU fused
//  - pool: f32 atomics into 512x128, then block-wise log_softmax
// ---------------------------------------------------------------------------

typedef __attribute__((ext_vector_type(16))) __bf16 v16bf;
typedef __attribute__((ext_vector_type(8)))  float  v8f;

#define NHID 128
#define APAD 136   // LDS row stride (elements): 272B -> conflict-free b128 lane access

// float -> bf16 bits, round-to-nearest-even
static __device__ __forceinline__ unsigned short f2bf(float f) {
    unsigned u = __builtin_bit_cast(unsigned, f);
    u += 0x7FFFu + ((u >> 16) & 1u);
    return (unsigned short)(u >> 16);
}

static __device__ __forceinline__ v16bf pack16(uint4 lo, uint4 hi) {
    union { uint4 u[2]; v16bf v; } t;
    t.u[0] = lo; t.u[1] = hi;
    return t.v;
}

// --------------------------- small utility kernels -------------------------

__global__ void gin_cvt_weights(const float* __restrict__ W1,
                                const float* __restrict__ W2,
                                unsigned short* __restrict__ out, int n_each) {
    int i = blockIdx.x * blockDim.x + threadIdx.x;
    if (i < n_each)           out[i] = f2bf(W1[i]);
    else if (i < 2 * n_each)  out[i] = f2bf(W2[i - n_each]);
}

__global__ void gin_copy4(const float4* __restrict__ src, float4* __restrict__ dst, int n4) {
    int i = blockIdx.x * blockDim.x + threadIdx.x;
    if (i < n4) dst[i] = src[i];
}

__global__ void gin_zero(float* __restrict__ p, int n) {
    int i = blockIdx.x * blockDim.x + threadIdx.x;
    if (i < n) p[i] = 0.f;
}

// one wave per edge, each lane moves 4 floats (float4 gather + 4 f32 atomics)
__global__ void gin_scatter(const float* __restrict__ h, const int* __restrict__ src,
                            const int* __restrict__ dst, float* __restrict__ z, int E) {
    long long gid = (long long)blockIdx.x * blockDim.x + threadIdx.x;
    int e = (int)(gid >> 5);
    if (e >= E) return;
    int c = ((int)gid & 31) * 4;
    long long s = src[e], d = dst[e];
    float4 v = *(const float4*)&h[s * NHID + c];
    float* zp = &z[d * NHID + c];
    atomicAdd(zp + 0, v.x);
    atomicAdd(zp + 1, v.y);
    atomicAdd(zp + 2, v.z);
    atomicAdd(zp + 3, v.w);
}

__global__ void gin_pool(const float* __restrict__ h, const int* __restrict__ batch,
                         float* __restrict__ pooled, int n) {
    int i = blockIdx.x * blockDim.x + threadIdx.x;
    if (i >= n) return;
    int node = i >> 7, c = i & (NHID - 1);
    atomicAdd(&pooled[(long long)batch[node] * NHID + c], h[i]);
}

__global__ void gin_log_softmax(const float* __restrict__ pooled, float* __restrict__ out) {
    __shared__ float red[NHID];
    int g = blockIdx.x, c = threadIdx.x;
    float v = pooled[g * NHID + c];
    red[c] = v; __syncthreads();
    for (int s = 64; s > 0; s >>= 1) { if (c < s) red[c] = fmaxf(red[c], red[c + s]); __syncthreads(); }
    float mx = red[0]; __syncthreads();
    red[c] = __expf(v - mx); __syncthreads();
    for (int s = 64; s > 0; s >>= 1) { if (c < s) red[c] += red[c + s]; __syncthreads(); }
    out[g * NHID + c] = v - (mx + __logf(red[0]));
}

// --------------------------- WMMA GEMM:  C = relu(A @ W^T + b) -------------
// A: [M,128] f32 (row-major), W(bf16): [128,128] row-major (W[n][k]), C: [M,128] f32.
// Block = 256 threads = 8 waves; block does 64 rows, wave w does N-cols [16w,16w+16).
// Safe for C == A (all global A reads staged to LDS before the barrier).
__global__ __launch_bounds__(256) void gin_gemm_relu(
    const float* __restrict__ A, const unsigned short* __restrict__ Wb,
    const float* __restrict__ bias, float* C, int M)
{
    __shared__ unsigned short As[64 * APAD];
    __shared__ unsigned short Ws[NHID * APAD];
    const int tid = threadIdx.x;
    const long long row0 = (long long)blockIdx.x * 64;

    // stage W: 128x128 bf16 (uint4 = 8 elems per load)
    #pragma unroll
    for (int i = 0; i < 8; ++i) {
        int idx = tid + i * 256;               // 0..2047
        int r = idx >> 4, c8 = (idx & 15) * 8;
        *(uint4*)&Ws[r * APAD + c8] = *(const uint4*)&Wb[r * NHID + c8];
    }
    // stage A: 64x128 f32 -> bf16 (float4 per load)
    #pragma unroll
    for (int i = 0; i < 8; ++i) {
        int idx = tid + i * 256;               // 0..2047
        int r = idx >> 5, c4 = (idx & 31) * 4;
        long long g = row0 + r;
        float4 v = {0.f, 0.f, 0.f, 0.f};
        if (g < M) v = *(const float4*)&A[g * NHID + c4];
        ushort4 p;
        p.x = f2bf(v.x); p.y = f2bf(v.y); p.z = f2bf(v.z); p.w = f2bf(v.w);
        *(ushort4*)&As[r * APAD + c4] = p;
    }
    __syncthreads();

    const int wave = tid >> 5;
    const int lane = tid & 31;
    const int half = lane >> 4;     // lane half selects K sub-range (ISA 7.12.2)
    const int l15  = lane & 15;
    const int ncol = wave * 16 + l15;

    // B fragments (32x16 per K-step): lanes 0-15 K=[0,16), lanes 16-31 K=[16,32)
    v16bf bfr[4];
    {
        const unsigned short* wrow = &Ws[ncol * APAD];
        #pragma unroll
        for (int k = 0; k < 4; ++k) {
            int kb = k * 32 + half * 16;
            bfr[k] = pack16(*(const uint4*)(wrow + kb), *(const uint4*)(wrow + kb + 8));
        }
    }
    const float bv = bias[ncol];

    #pragma unroll
    for (int mt = 0; mt < 4; ++mt) {
        long long rbase = row0 + mt * 16;
        if (rbase >= M) break;                  // M % 16 == 0, so subtile is all-or-nothing
        v8f acc = { bv, bv, bv, bv, bv, bv, bv, bv };   // bias folded into C
        const unsigned short* arow = &As[(mt * 16 + l15) * APAD];
        #pragma unroll
        for (int k = 0; k < 4; ++k) {
            // A 16x32 frag: elems 0..15 = K rel {half*16..+15}
            int ka = k * 32 + half * 16;
            v16bf a = pack16(*(const uint4*)(arow + ka), *(const uint4*)(arow + ka + 8));
            acc = __builtin_amdgcn_wmma_f32_16x16x32_bf16(
                false, a, false, bfr[k], (short)0, acc, false, false);
        }
        // D layout: VGPR r holds row (r + 8*half), col = l15; fuse ReLU
        float* crow = &C[(rbase + half * 8) * NHID + ncol];
        #pragma unroll
        for (int r = 0; r < 8; ++r) {
            float x = acc[r];
            crow[(long long)r * NHID] = x > 0.f ? x : 0.f;
        }
    }
}

// ------------------------------- launcher ----------------------------------

extern "C" void kernel_launch(void* const* d_in, const int* in_sizes, int n_in,
                              void* d_out, int out_size, void* d_ws, size_t ws_size,
                              hipStream_t stream) {
    const float* x     = (const float*)d_in[0];
    const int*   ei    = (const int*)d_in[1];
    const int*   batch = (const int*)d_in[2];
    const float* W1    = (const float*)d_in[3];
    const float* b1    = (const float*)d_in[4];
    const float* W2    = (const float*)d_in[5];
    const float* b2    = (const float*)d_in[6];
    float* out = (float*)d_out;

    const int N  = in_sizes[2];            // nodes
    const int E  = in_sizes[1] / 2;        // edges
    const int NL = in_sizes[4] / NHID;     // layers
    const int G  = out_size / NHID;        // graphs

    const int* src = ei;
    const int* dst = ei + E;

    // workspace: zbuf | hbuf | bf16 weights | pooled
    char* ws = (char*)d_ws;
    size_t nodeBytes = (size_t)N * NHID * sizeof(float);
    float*          zbuf = (float*)ws;
    float*          hbuf = (float*)(ws + nodeBytes);
    unsigned short* wb   = (unsigned short*)(ws + 2 * nodeBytes);
    size_t wbElems = (size_t)NL * NHID * NHID;           // per weight set
    size_t wbBytes = (2 * wbElems * sizeof(unsigned short) + 255) & ~(size_t)255;
    float*          pooled = (float*)(ws + 2 * nodeBytes + wbBytes);

    // 1) convert weights f32 -> bf16 once
    {
        int total = (int)(2 * wbElems);
        gin_cvt_weights<<<(total + 255) / 256, 256, 0, stream>>>(W1, W2, wb, (int)wbElems);
    }

    const int n4 = (N * NHID) / 4;
    const int gemmBlocks = (N + 63) / 64;
    const float* h = x;
    for (int l = 0; l < NL; ++l) {
        // z = h  (GIN eps=0 self term), then z += scatter-add of h[src] at dst
        gin_copy4<<<(n4 + 255) / 256, 256, 0, stream>>>((const float4*)h, (float4*)zbuf, n4);
        long long sthreads = (long long)E * 32;
        gin_scatter<<<(unsigned)((sthreads + 255) / 256), 256, 0, stream>>>(h, src, dst, zbuf, E);
        // z = relu(z @ W1^T + b1)  (in-place), h' = relu(z @ W2^T + b2)
        gin_gemm_relu<<<gemmBlocks, 256, 0, stream>>>(
            zbuf, wb + (size_t)l * NHID * NHID, b1 + l * NHID, zbuf, N);
        gin_gemm_relu<<<gemmBlocks, 256, 0, stream>>>(
            zbuf, wb + wbElems + (size_t)l * NHID * NHID, b2 + l * NHID, hbuf, N);
        h = hbuf;
    }

    // global_add_pool + log_softmax
    gin_zero<<<(G * NHID + 255) / 256, 256, 0, stream>>>(pooled, G * NHID);
    gin_pool<<<(N * NHID + 255) / 256, 256, 0, stream>>>(h, batch, pooled, N * NHID);
    gin_log_softmax<<<G, NHID, 0, stream>>>(pooled, out);
}
